// FlashAttention_10144712753257
// MI455X (gfx1250) — matrix operand order repaired and checked by reading, not compiled, not run on hardware
//
#include <hip/hip_runtime.h>
#include <hip/hip_bf16.h>

typedef __attribute__((ext_vector_type(16))) _Float16 v16h;
typedef __attribute__((ext_vector_type(4)))  _Float16 v4h;
typedef __attribute__((ext_vector_type(8)))  float    v8f;

#define B_   2
#define L_   2048
#define H_   16
#define D_   64
#define KVB  128          // KV block (matches reference KV_BLOCK for epsilon term)
#define QB   128          // queries per workgroup (8 waves x 16 rows)
#define KROW 72           // 64 + 8 pad halves  (row stride 144B, 16B aligned, conflict-free frag reads)
#define VROW 136          // 128 + 8 pad halves (row stride 272B, 16B aligned, conflict-free frag reads)

__launch_bounds__(256, 2)
__global__ void fa_fwd_f16wmma(const float* __restrict__ Q, const float* __restrict__ K,
                               const float* __restrict__ V, float* __restrict__ Out) {
  __shared__ _Float16 ldsK[KVB * KROW];   // K block, row-major [key][d]
  __shared__ _Float16 ldsV[D_  * VROW];   // V block, transposed [d][key]

  const int qblk = blockIdx.x & 15;
  const int h    = (blockIdx.x >> 4) & 15;
  const int b    = blockIdx.x >> 8;
  const int tid  = threadIdx.x;
  const int wave = tid >> 5;
  const int lane = tid & 31;
  const int lh   = lane & 15;
  const bool hi  = lane >= 16;

  const int qbase = qblk * QB + wave * 16;

  // per-(b,h) base pointers; key stride in global memory is H_*D_ floats
  const float* Kbh = K + (((size_t)b * L_) * H_ + h) * D_;
  const float* Vbh = V + (((size_t)b * L_) * H_ + h) * D_;

  // ---- Q as WMMA B-fragments (pre-scaled by 1/sqrt(D)); lane = query column ----
  // B operand, contiguous k order (matches A): lanes 0-15 hold K=0..15 (N=lane), lanes 16-31 hold K=16..31 (N=lane-16)
  v16h qb[2];
  {
    const float* qp = Q + (((size_t)b * L_ + (qbase + lh)) * H_ + h) * D_;
#pragma unroll
    for (int c = 0; c < 2; ++c) {
      const float* p = qp + c * 32 + (hi ? 16 : 0);
#pragma unroll
      for (int j = 0; j < 16; ++j) qb[c][j] = (_Float16)(p[j] * 0.125f);
    }
  }

  v8f acc[4];                 // O tile 16q x 64d: 4 dim-tiles of 16, f32 accum
#pragma unroll
  for (int n = 0; n < 4; ++n)
#pragma unroll
    for (int j = 0; j < 8; ++j) acc[n][j] = 0.0f;

  float m = -__builtin_huge_valf();   // running row max (per lane = per query)
  float l = 0.0f;                     // running denominator

  for (int kv = 0; kv < L_ / KVB; ++kv) {
    const int kvbase = kv * KVB;

    // ---- cooperative stage: K row-major f16, V transposed f16 ----
    // Each thread owns 8 chunks of 4 consecutive dims -> float4 global loads,
    // packed v4h (b64) stores for K; 4x b16 scatter for the V transpose.
#pragma unroll
    for (int i = 0; i < (KVB * D_) / (256 * 4); ++i) {   // 8 iters
      const int c   = tid + i * 256;                     // chunk id 0..2047
      const int row = c >> 4;                            // key in block
      const int col = (c & 15) << 2;                     // dim (multiple of 4)
      const size_t g = (size_t)(kvbase + row) * (H_ * D_) + col;
      const float4 kq = *(const float4*)(Kbh + g);
      const float4 vq = *(const float4*)(Vbh + g);
      v4h kh;
      kh[0] = (_Float16)kq.x; kh[1] = (_Float16)kq.y;
      kh[2] = (_Float16)kq.z; kh[3] = (_Float16)kq.w;
      *(v4h*)&ldsK[row * KROW + col] = kh;
      ldsV[(col + 0) * VROW + row] = (_Float16)vq.x;
      ldsV[(col + 1) * VROW + row] = (_Float16)vq.y;
      ldsV[(col + 2) * VROW + row] = (_Float16)vq.z;
      ldsV[(col + 3) * VROW + row] = (_Float16)vq.w;
    }
    __syncthreads();

    // ---- prefetch next KV block into cache while this block computes ----
    if (kv + 1 < L_ / KVB) {
      const int prow  = tid >> 1;                         // 128 rows, 2 lines each
      const int phalf = (tid & 1) * 32;                   // 128B halves of the 256B row
      const size_t pg = (size_t)(kvbase + KVB + prow) * (H_ * D_) + phalf;
      __builtin_prefetch(Kbh + pg, 0, 3);
      __builtin_prefetch(Vbh + pg, 0, 3);
    }

    // ---- S^T tiles: st[t] = (K tile t) x Q  -> lane = query, vgpr idx = key ----
    v8f st[8];
#pragma unroll
    for (int t = 0; t < 8; ++t)
#pragma unroll
      for (int j = 0; j < 8; ++j) st[t][j] = 0.0f;

#pragma unroll
    for (int t = 0; t < 8; ++t) {
      const _Float16* krow = &ldsK[(t * 16 + lh) * KROW];
#pragma unroll
      for (int c = 0; c < 2; ++c) {
        // A operand, contiguous k order (matches Q): lanes<16: K=c*32+{0..15}; lanes>=16: +16
        const int base = c * 32 + (hi ? 16 : 0);
        v16h a;
#pragma unroll
        for (int j = 0; j < 16; ++j) a[j] = krow[base + j];
        st[t] = __builtin_amdgcn_wmma_f32_16x16x32_f16(false, a, false, qb[c],
                                                       (short)0, st[t], false, false);
      }
    }

    // ---- per-query block max (in-lane over 8x8 + one cross-half combine) ----
    float mb = st[0][0];
#pragma unroll
    for (int t = 0; t < 8; ++t)
#pragma unroll
      for (int j = 0; j < 8; ++j) mb = fmaxf(mb, st[t][j]);
    mb = fmaxf(mb, __shfl_xor(mb, 16, 32));

    const float mnew  = fmaxf(m, mb);
    const float alpha = __expf(m - mnew);

    // ---- P = exp(S - mnew); row sum ----
    float rs = 0.0f;
#pragma unroll
    for (int t = 0; t < 8; ++t)
#pragma unroll
      for (int j = 0; j < 8; ++j) {
        const float p = __expf(st[t][j] - mnew);
        st[t][j] = p;
        rs += p;
      }
    rs += __shfl_xor(rs, 16, 32);
    l = l * alpha + rs + 1e-10f * __expf(mb - mnew);   // faithful per-block epsilon
    m = mnew;

    // ---- rescale O by per-row alpha (O rows: M = r + (hi?8:0)) ----
    {
      const int rq = hi ? 8 : 0;
#pragma unroll
      for (int r = 0; r < 8; ++r) {
        const float ar = __shfl(alpha, r + rq, 32);
#pragma unroll
        for (int n = 0; n < 4; ++n) acc[n][r] *= ar;
      }
    }

    // ---- P -> A-fragments: pure in-lane f32->f16 pack of adjacent S^T tiles ----
    v16h pa[4];
#pragma unroll
    for (int kk = 0; kk < 4; ++kk)
#pragma unroll
      for (int j = 0; j < 8; ++j) {
        pa[kk][j]     = (_Float16)st[2 * kk][j];
        pa[kk][8 + j] = (_Float16)st[2 * kk + 1][j];
      }

    // ---- O += P x V  (B-fragments from transposed V: contiguous 16B LDS reads) ----
#pragma unroll
    for (int n = 0; n < 4; ++n) {
      const _Float16* vrow = &ldsV[(n * 16 + lh) * VROW];
#pragma unroll
      for (int kk = 0; kk < 4; ++kk) {
        const int base = kk * 32 + (hi ? 8 : 0);
        v16h bf;
#pragma unroll
        for (int j = 0; j < 16; ++j) bf[j] = vrow[base + j + (j >= 8 ? 8 : 0)];
        acc[n] = __builtin_amdgcn_wmma_f32_16x16x32_f16(false, pa[kk], false, bf,
                                                        (short)0, acc[n], false, false);
      }
    }
    __syncthreads();   // protect LDS before next block's staging
  }

  // ---- finalize: divide by l, store [B,L,H,D] ----
  {
    const int rq = hi ? 8 : 0;
#pragma unroll
    for (int r = 0; r < 8; ++r) {
      const float inv = 1.0f / __shfl(l, r + rq, 32);
      const int qrow = qbase + r + rq;
      float* op = Out + (((size_t)b * L_ + qrow) * H_ + h) * D_ + lh;
#pragma unroll
      for (int n = 0; n < 4; ++n) op[n * 16] = acc[n][r] * inv;
    }
  }
}

extern "C" void kernel_launch(void* const* d_in, const int* in_sizes, int n_in,
                              void* d_out, int out_size, void* d_ws, size_t ws_size,
                              hipStream_t stream) {
  const float* q = (const float*)d_in[0];
  const float* k = (const float*)d_in[1];
  const float* v = (const float*)d_in[2];
  float* out = (float*)d_out;
  (void)in_sizes; (void)n_in; (void)out_size; (void)d_ws; (void)ws_size;

  dim3 grid(B_ * H_ * (L_ / QB));   // 512 workgroups
  dim3 block(256);                  // 8 wave32 waves
  fa_fwd_f16wmma<<<grid, block, 0, stream>>>(q, k, v, out);
}
